// ResGatedGCN1_17386027614851
// MI455X (gfx1250) — matrix operand order repaired and checked by reading, not compiled, not run on hardware
//
#include <hip/hip_runtime.h>
#include <hip/hip_bf16.h>

// ---------------------------------------------------------------------------
// Types for CDNA5 WMMA (wave32): v_wmma_f32_16x16x32_bf16
// ---------------------------------------------------------------------------
typedef __attribute__((ext_vector_type(16))) __bf16 bfx16;
typedef __attribute__((ext_vector_type(8)))  __bf16 bfx8;
typedef __attribute__((ext_vector_type(8)))  float  fx8;

union ABUnion { bfx16 v; bfx8 h[2]; };

// Problem constants (match reference)
#define NN 50000
#define EE 800000
#define INDIM 64
#define HID 128

__device__ __forceinline__ void atomAddF32(float* p, float v) {
    unsafeAtomicAdd(p, v);   // lowers to global_atomic_add_f32
}

// A fragment, 16x32 bf16 (MxK). Lane l (0..15): row m0+l, K = kk+0..7 (v0..3),
// kk+16..23 (v4..7). Lane l+16: same row, K offsets +8 / +24. Pairs contiguous.
__device__ __forceinline__ bfx16 load_a_frag(const __bf16* rowptr, int kk, int hi8) {
    ABUnion u;
    u.h[0] = *reinterpret_cast<const bfx8*>(rowptr + kk + hi8);
    u.h[1] = *reinterpret_cast<const bfx8*>(rowptr + kk + 16 + hi8);
    return u.v;
}

// B fragment, 32x16 bf16 (KxN), B[k][n] = W[n][k], W row-major [NOUT,K].
// Lane l holds column n = (l&15); lanes 0..15 -> K=kk+0..7, kk+16..23; 16..31 -> +8.
// Reads from LDS-staged weights -> ds_load path.
__device__ __forceinline__ bfx16 load_b_frag_lds(const __bf16* wrow, int kk, int hi16) {
    ABUnion u;
    u.h[0] = *reinterpret_cast<const bfx8*>(wrow + kk + (hi16 >> 1));
    u.h[1] = *reinterpret_cast<const bfx8*>(wrow + kk + 16 + (hi16 >> 1));
    return u.v;
}

// ---------------------------------------------------------------------------
// f32 -> bf16 conversion (grid-stride)
// ---------------------------------------------------------------------------
__global__ void cvt_f32_bf16(const float* __restrict__ in, __bf16* __restrict__ out, int n) {
    int i = blockIdx.x * blockDim.x + threadIdx.x;
    int stride = gridDim.x * blockDim.x;
    for (; i < n; i += stride) out[i] = (__bf16)in[i];
}

// ---------------------------------------------------------------------------
// e0 = edge_w * emb_e_w + emb_e_b   (rank-1 "GEMM", [E,64] bf16 out)
// ---------------------------------------------------------------------------
__global__ void edge_embed_kernel(const float* __restrict__ ew,
                                  const float* __restrict__ W,   // [64]
                                  const float* __restrict__ b,   // [64]
                                  __bf16* __restrict__ e0) {
    long long idx = (long long)blockIdx.x * blockDim.x + threadIdx.x;
    if (idx >= (long long)EE * 64) return;
    int k = (int)(idx >> 6), c = (int)(idx & 63);
    e0[idx] = (__bf16)(ew[k] * W[c] + b[c]);
}

// ---------------------------------------------------------------------------
// In-degree (computed once; dst is layer-invariant)
// ---------------------------------------------------------------------------
__global__ void deg_kernel(const int* __restrict__ dst, float* __restrict__ deg) {
    int k = blockIdx.x * blockDim.x + threadIdx.x;
    if (k < EE) atomAddF32(&deg[dst[k]], 1.0f);
}

// ---------------------------------------------------------------------------
// Generic WMMA GEMM: out[M,NOUT] = A[M,K](bf16) @ W[NOUT,K]^T (bf16) + bias
// Grid: (ceil(Mtiles/8), NOUT/64). All 8 waves of a block share one 64-row
// weight slice, cooperatively staged in LDS (16KB max). One wave = 16x64 tile.
// ---------------------------------------------------------------------------
template<int K, int NOUT, bool OUT_BF16>
__global__ __launch_bounds__(256) void wmma_gemm_kernel(
        const __bf16* __restrict__ A,
        const __bf16* __restrict__ W,
        const float* __restrict__ bias,
        void* __restrict__ out, int M) {
    __shared__ __align__(32) __bf16 sW[64 * K];

    int nt = blockIdx.y;
    int n0 = nt << 6;
    // Cooperative stage of W rows [n0, n0+64) x K (bf16) into LDS.
    for (int i = threadIdx.x * 8; i < 64 * K; i += 256 * 8)
        *reinterpret_cast<bfx8*>(&sW[i]) =
            *reinterpret_cast<const bfx8*>(&W[(size_t)n0 * K + i]);
    __syncthreads();

    int mt = blockIdx.x * 8 + (threadIdx.x >> 5);
    if (mt >= (M >> 4)) return;
    int lane = threadIdx.x & 31;
    int m0 = mt << 4;
    int li = lane & 15, hi = lane >> 4;
    int hi8 = hi << 3, hi16 = hi << 4;

    fx8 acc[4];
#pragma unroll
    for (int t = 0; t < 4; ++t) {
        float b = bias[n0 + t * 16 + li];
#pragma unroll
        for (int i = 0; i < 8; ++i) acc[t][i] = b;
    }

    const __bf16* arow = A + (size_t)(m0 + li) * K;
#pragma unroll
    for (int kk = 0; kk < K; kk += 32) {
        bfx16 a = load_a_frag(arow, kk, hi8);
#pragma unroll
        for (int t = 0; t < 4; ++t) {
            bfx16 b = load_b_frag_lds(&sW[(t * 16 + li) * K], kk, hi16);
            acc[t] = __builtin_amdgcn_wmma_f32_16x16x32_bf16(
                false, a, false, b, (short)0, acc[t], false, false);
        }
    }

#pragma unroll
    for (int t = 0; t < 4; ++t) {
        int n = n0 + t * 16 + li;
#pragma unroll
        for (int v = 0; v < 8; ++v) {
            int m = m0 + v + hi8;
            float val = acc[t][v];
            if (OUT_BF16) ((__bf16*)out)[(size_t)m * NOUT + n] = (__bf16)val;
            else          ((float*)out)[(size_t)m * NOUT + n] = val;
        }
    }
}

// ---------------------------------------------------------------------------
// Fused edge kernel: per 16-edge tile one wave computes
//   Ce = e @ C_w^T + C_b   (WMMA, all 128 output cols = 8 tiles)
//   e_new = Ce + Ah[dst] + Bh[src];  sig = sigmoid(e_new)
//   atomic num[dst] += Vh[src]*sig;  den[dst] += sig
//   e_out = bf16(relu(e_new))        (in place when K==128)
// C_w staged in LDS (16/32KB per block, shared by 8 waves) so the VMEM pipe
// is left for the UVAB gathers and num/den atomics.
// UVAB layout: [N,512] f32 with col blocks U:0 V:128 A:256 B:384.
// ---------------------------------------------------------------------------
template<int K>
__global__ __launch_bounds__(256) void edge_layer_kernel(
        const __bf16* Ein, __bf16* Eout,
        const __bf16* __restrict__ WC,
        const float* __restrict__ bC,
        const float* __restrict__ UVAB,
        const int* __restrict__ src,
        const int* __restrict__ dst,
        float* __restrict__ num,
        float* __restrict__ den,
        int storeE) {
    __shared__ __align__(32) __bf16 sWC[128 * K];

    for (int i = threadIdx.x * 8; i < 128 * K; i += 256 * 8)
        *reinterpret_cast<bfx8*>(&sWC[i]) =
            *reinterpret_cast<const bfx8*>(&WC[i]);
    __syncthreads();

    int wave = blockIdx.x * 8 + (threadIdx.x >> 5);   // exactly EE/16 waves
    int lane = threadIdx.x & 31;
    int m0 = wave << 4;
    int li = lane & 15, hi = lane >> 4;
    int hi8 = hi << 3, hi16 = hi << 4;

    fx8 acc[8];
#pragma unroll
    for (int t = 0; t < 8; ++t) {
        float b = bC[t * 16 + li];
#pragma unroll
        for (int i = 0; i < 8; ++i) acc[t][i] = b;
    }

    const __bf16* arow = Ein + (size_t)(m0 + li) * K;
#pragma unroll
    for (int kk = 0; kk < K; kk += 32) {
        bfx16 a = load_a_frag(arow, kk, hi8);
#pragma unroll
        for (int t = 0; t < 8; ++t) {
            bfx16 b = load_b_frag_lds(&sWC[(t * 16 + li) * K], kk, hi16);
            acc[t] = __builtin_amdgcn_wmma_f32_16x16x32_bf16(
                false, a, false, b, (short)0, acc[t], false, false);
        }
    }

    int srows[8], drows[8];
#pragma unroll
    for (int v = 0; v < 8; ++v) {
        int m = m0 + v + hi8;
        srows[v] = src[m];
        drows[v] = dst[m];
    }

#pragma unroll
    for (int t = 0; t < 8; ++t) {
        int n = t * 16 + li;
#pragma unroll
        for (int v = 0; v < 8; ++v) {
            int m = m0 + v + hi8;
            int s = srows[v], d = drows[v];
            float enew = acc[t][v]
                       + UVAB[(size_t)d * 512 + 256 + n]   // Ah[dst]
                       + UVAB[(size_t)s * 512 + 384 + n];  // Bh[src]
            float sig = 1.0f / (1.0f + __expf(-enew));
            float vh  = UVAB[(size_t)s * 512 + 128 + n];   // Vh[src]
            atomAddF32(&num[(size_t)d * 128 + n], vh * sig);
            atomAddF32(&den[(size_t)d * 128 + n], sig);
            if (storeE) Eout[(size_t)m * 128 + n] = (__bf16)fmaxf(enew, 0.0f);
        }
    }
}

// ---------------------------------------------------------------------------
// Node update: h = relu(deg>0 ? Uh + num/(den+eps) : 0); optional pooling.
// ---------------------------------------------------------------------------
__global__ void node_update_kernel(const float* __restrict__ UVAB,
                                   const float* __restrict__ num,
                                   const float* __restrict__ den,
                                   const float* __restrict__ deg,
                                   __bf16* __restrict__ hout,
                                   const int* __restrict__ batch,
                                   float* __restrict__ psum,
                                   float* __restrict__ pcnt,
                                   int pool) {
    int idx = blockIdx.x * blockDim.x + threadIdx.x;
    if (idx >= NN * HID) return;
    int n = idx >> 7, c = idx & 127;
    float val = 0.0f;
    if (deg[n] > 0.0f) {
        val = UVAB[(size_t)n * 512 + c] + num[idx] / (den[idx] + 1e-16f);
    }
    val = fmaxf(val, 0.0f);
    hout[idx] = (__bf16)val;
    if (pool) {
        int g = batch[n];
        atomAddF32(&psum[g * HID + c], val);
        if (c == 0) atomAddF32(&pcnt[g], 1.0f);
    }
}

__global__ void pool_finalize_kernel(const float* __restrict__ psum,
                                     const float* __restrict__ pcnt,
                                     float* __restrict__ out) {
    int idx = blockIdx.x * blockDim.x + threadIdx.x;
    if (idx >= 16 * HID) return;
    int g = idx >> 7;
    out[idx] = psum[idx] / fmaxf(pcnt[g], 1.0f);
}

// ---------------------------------------------------------------------------
// Host launcher
// ---------------------------------------------------------------------------
extern "C" void kernel_launch(void* const* d_in, const int* in_sizes, int n_in,
                              void* d_out, int out_size, void* d_ws, size_t ws_size,
                              hipStream_t stream) {
    (void)in_sizes; (void)n_in; (void)out_size; (void)ws_size;

    // Inputs in setup_inputs() insertion order.
    const float* x       = (const float*)d_in[0];
    const float* edge_w  = (const float*)d_in[1];
    const int*   eidx    = (const int*)d_in[2];
    const int*   src     = eidx;          // edge_index[0]
    const int*   dst     = eidx + EE;     // edge_index[1]
    const int*   batch   = (const int*)d_in[3];
    const float* emb_h_w = (const float*)d_in[4];
    const float* emb_h_b = (const float*)d_in[5];
    const float* emb_e_w = (const float*)d_in[6];
    const float* emb_e_b = (const float*)d_in[7];

    const float* Uw[4], *Ub[4], *Vw[4], *Vb[4], *Aw[4], *Ab[4], *Bw[4], *Bb[4], *Cw[4], *Cb[4];
    for (int l = 0; l < 4; ++l) {
        int base = 8 + l * 10;
        Uw[l] = (const float*)d_in[base + 0]; Ub[l] = (const float*)d_in[base + 1];
        Vw[l] = (const float*)d_in[base + 2]; Vb[l] = (const float*)d_in[base + 3];
        Aw[l] = (const float*)d_in[base + 4]; Ab[l] = (const float*)d_in[base + 5];
        Bw[l] = (const float*)d_in[base + 6]; Bb[l] = (const float*)d_in[base + 7];
        Cw[l] = (const float*)d_in[base + 8]; Cb[l] = (const float*)d_in[base + 9];
    }

    // Workspace carve-up (256B aligned).
    char* ws = (char*)d_ws;
    size_t off = 0;
    auto alloc = [&](size_t bytes) -> char* {
        char* p = ws + off;
        off = (off + bytes + 255) & ~(size_t)255;
        return p;
    };
    __bf16* xbf   = (__bf16*)alloc((size_t)NN * 64 * 2);
    __bf16* hA    = (__bf16*)alloc((size_t)NN * HID * 2);   // also holds h0 [N,64]
    __bf16* hB    = (__bf16*)alloc((size_t)NN * HID * 2);
    __bf16* e0    = (__bf16*)alloc((size_t)EE * 64 * 2);
    __bf16* eA    = (__bf16*)alloc((size_t)EE * HID * 2);   // in-place for layers 1..3
    float*  UVAB  = (float*) alloc((size_t)NN * 512 * 4);
    float*  num   = (float*) alloc((size_t)NN * HID * 4);
    float*  den   = (float*) alloc((size_t)NN * HID * 4);
    float*  deg   = (float*) alloc((size_t)NN * 4);
    float*  psum  = (float*) alloc(16 * HID * 4);
    float*  pcnt  = (float*) alloc(16 * 4);
    __bf16* WbEmb = (__bf16*)alloc(64 * 64 * 2);
    __bf16* Wstk[4]; float* bstk[4]; __bf16* WCb[4];
    for (int l = 0; l < 4; ++l) {
        int K = (l == 0) ? 64 : 128;
        Wstk[l] = (__bf16*)alloc((size_t)512 * K * 2);
        bstk[l] = (float*) alloc(512 * 4);
        WCb[l]  = (__bf16*)alloc((size_t)128 * K * 2);
    }

    auto cvt = [&](const float* in, __bf16* out, int n) {
        int blocks = (n + 255) / 256; if (blocks > 2048) blocks = 2048;
        cvt_f32_bf16<<<blocks, 256, 0, stream>>>(in, out, n);
    };

    // --- weight / input conversion to bf16 ---
    cvt(x, xbf, NN * 64);
    cvt(emb_h_w, WbEmb, 64 * 64);
    for (int l = 0; l < 4; ++l) {
        int K = (l == 0) ? 64 : 128;
        cvt(Uw[l], Wstk[l] + (size_t)0   * K, 128 * K);
        cvt(Vw[l], Wstk[l] + (size_t)128 * K, 128 * K);
        cvt(Aw[l], Wstk[l] + (size_t)256 * K, 128 * K);
        cvt(Bw[l], Wstk[l] + (size_t)384 * K, 128 * K);
        cvt(Cw[l], WCb[l], 128 * K);
        hipMemcpyAsync(bstk[l] + 0,   Ub[l], 128 * 4, hipMemcpyDeviceToDevice, stream);
        hipMemcpyAsync(bstk[l] + 128, Vb[l], 128 * 4, hipMemcpyDeviceToDevice, stream);
        hipMemcpyAsync(bstk[l] + 256, Ab[l], 128 * 4, hipMemcpyDeviceToDevice, stream);
        hipMemcpyAsync(bstk[l] + 384, Bb[l], 128 * 4, hipMemcpyDeviceToDevice, stream);
    }

    // --- embeddings ---
    {   // h0 = x @ emb_h_w^T + emb_h_b  (bf16 out, [N,64])
        dim3 grid((NN / 16 + 7) / 8, 64 / 64);
        wmma_gemm_kernel<64, 64, true><<<grid, 256, 0, stream>>>(
            xbf, WbEmb, emb_h_b, hA, NN);
    }
    {   // e0 = edge_w * emb_e_w + emb_e_b
        long long tot = (long long)EE * 64;
        edge_embed_kernel<<<(int)((tot + 255) / 256), 256, 0, stream>>>(
            edge_w, emb_e_w, emb_e_b, e0);
    }

    // --- degree (layer invariant) + pooling accumulators ---
    hipMemsetAsync(deg, 0, (size_t)NN * 4, stream);
    deg_kernel<<<(EE + 255) / 256, 256, 0, stream>>>(dst, deg);
    hipMemsetAsync(psum, 0, 16 * HID * 4, stream);
    hipMemsetAsync(pcnt, 0, 16 * 4, stream);

    __bf16* h_cur = hA;
    __bf16* h_nxt = hB;

    for (int l = 0; l < 4; ++l) {
        int K = (l == 0) ? 64 : 128;
        hipMemsetAsync(num, 0, (size_t)NN * HID * 4, stream);
        hipMemsetAsync(den, 0, (size_t)NN * HID * 4, stream);

        // Fused U/V/A/B node GEMM -> UVAB [N,512] f32
        dim3 ggrid((NN / 16 + 7) / 8, 512 / 64);
        if (K == 64)
            wmma_gemm_kernel<64, 512, false><<<ggrid, 256, 0, stream>>>(
                h_cur, Wstk[l], bstk[l], UVAB, NN);
        else
            wmma_gemm_kernel<128, 512, false><<<ggrid, 256, 0, stream>>>(
                h_cur, Wstk[l], bstk[l], UVAB, NN);

        // Fused edge kernel: exactly (EE/16)/8 = 6250 full blocks.
        int eblocks = (EE / 16) / 8;
        int storeE = (l == 3) ? 0 : 1;
        if (K == 64)
            edge_layer_kernel<64><<<eblocks, 256, 0, stream>>>(
                e0, eA, WCb[l], Cb[l], UVAB, src, dst, num, den, storeE);
        else
            edge_layer_kernel<128><<<eblocks, 256, 0, stream>>>(
                eA, eA, WCb[l], Cb[l], UVAB, src, dst, num, den, storeE);

        // Node update (+ fused mean-pool accumulation on last layer)
        node_update_kernel<<<(NN * HID + 255) / 256, 256, 0, stream>>>(
            UVAB, num, den, deg, h_nxt, batch, psum, pcnt, (l == 3) ? 1 : 0);

        __bf16* t = h_cur; h_cur = h_nxt; h_nxt = t;
    }

    pool_finalize_kernel<<<(16 * HID + 255) / 256, 256, 0, stream>>>(
        psum, pcnt, (float*)d_out);
}
